// AutoregressiveDescriptor_32547262169128
// MI455X (gfx1250) — matrix operand order repaired and checked by reading, not compiled, not run on hardware
//
#include <hip/hip_runtime.h>

typedef __attribute__((ext_vector_type(16))) __bf16 v16bf;
typedef __attribute__((ext_vector_type(8)))  float  v8f;

#define WMMA_BF16(a, b, c) \
  __builtin_amdgcn_wmma_f32_16x16x32_bf16(false, (a), false, (b), (short)0, (c), false, false)

static __device__ __forceinline__ __bf16 to_bf(float x) { return (__bf16)x; }

// Truncate a generic pointer to an LDS byte offset (low 32 bits of the address, ISA 10.2).
static __device__ __forceinline__ unsigned lds_off(const void* p) {
  return (unsigned)(unsigned long long)p;
}

// ---------------------------------------------------------------------------
// fp32 -> bf16 weight pre-conversion (run once per launch; weights ~15 MB bf16,
// fully L2-resident afterwards).
// ---------------------------------------------------------------------------
__global__ void f32_to_bf16(const float* __restrict__ s, __bf16* __restrict__ d, int n) {
  int i = blockIdx.x * 256 + threadIdx.x;
  if (i < n) d[i] = (__bf16)s[i];
}

// ---------------------------------------------------------------------------
// GEMM: C[M,N] = act(A[M,K](f32) @ W[N,K](bf16)^T + bias[N] (+ res[M,N]))
// Block tile 64(M) x 128(N) x 32(K), 256 threads = 8 waves.
// Wave tile 16(M) x 64(N): 4 WMMAs per K-step with A-fragment reuse.
//
// Double-buffered software pipeline: B tile for step i+1 is DMA'd into the
// spare LDS buffer with global_load_async_to_lds_b128 (ASYNCcnt) while the
// WMMAs for step i run; A tile i+1 is converted fp32->bf16 in the same
// shadow. One s_wait_asynccnt + one barrier per K-step.
// M % 64 == 0, N % 128 == 0, K % 32 == 0 for every call in this model.
// perm!=0 remaps output row m=(t*64+b) -> b*16+t (final logits layout).
// ---------------------------------------------------------------------------
__global__ __launch_bounds__(256) void gemm_bf16_wmma(
    const float* __restrict__ A, const __bf16* __restrict__ W,
    const float* __restrict__ bias, const float* __restrict__ res,
    float* __restrict__ C, int M, int N, int K, int relu, int perm)
{
  __shared__ __bf16 As[2][64][36];
  __shared__ __attribute__((aligned(16))) __bf16 Bs[2][128][40]; // 80B rows: 16B-aligned
  const int tid  = threadIdx.x;
  const int lane = tid & 31;
  const int wave = tid >> 5;
  const int m0 = blockIdx.y << 6;
  const int n0 = blockIdx.x << 7;
  const int wm = (wave & 3) << 4;      // 0,16,32,48
  const int wn = (wave >> 2) << 6;     // 0,64
  const int lr = tid >> 2;             // A tile row (0..63)
  const int lc = (tid & 3) << 3;       // A tile col (8 floats)
  const int br = tid >> 1;             // B tile row (0..127)
  const int bc = (tid & 1) << 4;       // B tile col in bf16 (0,16) -> 32B chunk

  v8f acc[4];
  v8f zero = {};
  acc[0] = zero; acc[1] = zero; acc[2] = zero; acc[3] = zero;

  const float*  ap = A + (size_t)(m0 + lr) * K + lc;
  const __bf16* wp = W + (size_t)(n0 + br) * K + bc;

  // Stage one K-tile into LDS buffer `sb`: B via async DMA, A via VALU convert.
  auto stage = [&](int k0, int sb) {
    const __bf16* wpt = wp + k0;
    unsigned ldsb = lds_off(&Bs[sb][br][bc]);
    asm volatile("global_load_async_to_lds_b128 %0, %1, off"
                 :: "v"(ldsb), "v"(wpt) : "memory");
    asm volatile("global_load_async_to_lds_b128 %0, %1, off offset:16"
                 :: "v"(ldsb), "v"(wpt) : "memory");
#pragma unroll
    for (int j = 0; j < 8; ++j) As[sb][lr][lc + j] = to_bf(ap[k0 + j]);
    if (k0 + 32 < K) __builtin_prefetch(ap + k0 + 32, 0, 1);
  };

  stage(0, 0);
  int buf = 0;
  for (int k0 = 0; k0 < K; k0 += 32) {
    asm volatile("s_wait_asynccnt 0x0" ::: "memory");  // tile i DMA done (this wave)
    __syncthreads();                                   // all staging visible; prev compute retired
    if (k0 + 32 < K) stage(k0 + 32, buf ^ 1);          // overlap next DMA with this compute

    // A fragment: 16x32 bf16 (contiguous: lane<16 K-base 0, lane>=16 K-base 16)
    const int am  = wm + (lane & 15);
    const int akb = (lane < 16) ? 0 : 16;
    v16bf a;
#pragma unroll
    for (int e = 0; e < 8; ++e) {
      a[e]     = As[buf][am][akb + e];
      a[e + 8] = As[buf][am][8 + akb + e];
    }
    // B fragments: 32x16 bf16, n = lane&15, K-base 0/16 by lane half.
    // Load all four before the WMMAs so the DS loads clause and the four
    // WMMAs issue back-to-back.
    const int bkb = (lane < 16) ? 0 : 16;
    const int bn  = wn + (lane & 15);
    v16bf b0v, b1v, b2v, b3v;
#pragma unroll
    for (int e = 0; e < 16; ++e) {
      b0v[e] = Bs[buf][bn][bkb + e];
      b1v[e] = Bs[buf][bn + 16][bkb + e];
      b2v[e] = Bs[buf][bn + 32][bkb + e];
      b3v[e] = Bs[buf][bn + 48][bkb + e];
    }
    acc[0] = WMMA_BF16(a, b0v, acc[0]);
    acc[1] = WMMA_BF16(a, b1v, acc[1]);
    acc[2] = WMMA_BF16(a, b2v, acc[2]);
    acc[3] = WMMA_BF16(a, b3v, acc[3]);
    buf ^= 1;
  }

  const int nl = lane & 15;
  const int mh = (lane < 16) ? 0 : 8;
#pragma unroll
  for (int r = 0; r < 8; ++r) {
    int m = m0 + wm + mh + r;
    size_t row = perm ? (size_t)((m & 63) * 16 + (m >> 6)) : (size_t)m;
#pragma unroll
    for (int s = 0; s < 4; ++s) {
      int n = n0 + wn + 16 * s + nl;
      float v = acc[s][r] + bias[n];
      if (res)  v += res[(size_t)m * N + n];
      if (relu) v = fmaxf(v, 0.f);
      C[row * N + n] = v;
    }
  }
}

// ---------------------------------------------------------------------------
// Fused attention for one (batch b, head h, 16-row Q tile).  dh = 64.
// Scores and P@V on WMMA; fp32 softmax over full Skv (no mask, faithful).
// Ragged Sq/Skv handled by zero padding.
// ---------------------------------------------------------------------------
__global__ __launch_bounds__(32) void attn_wmma(
    const float* __restrict__ Q, long long qb, int qr, int qc,
    const float* __restrict__ Kp, long long kb, int kr, int kc,
    const float* __restrict__ V, long long vb, int vr, int vc,
    float* __restrict__ O, long long ob, int orr,
    int Sq, int Skv, float scale)
{
  __shared__ __bf16 Qs[16][68];
  __shared__ __bf16 Ks[16][68];
  __shared__ __bf16 Vs[32][68];
  __shared__ float  Ps[16][260];

  const int lane = threadIdx.x;
  const int h = blockIdx.y, b = blockIdx.z, qt = blockIdx.x;
  const float* Qb = Q  + (long long)b * qb + qc + h * 64;
  const float* Kb = Kp + (long long)b * kb + kc + h * 64;
  const float* Vb = V  + (long long)b * vb + vc + h * 64;

  const int r  = lane & 15;
  const int hf = lane >> 4;
  const int akb = (lane < 16) ? 0 : 16;
  const int bkb = (lane < 16) ? 0 : 16;
  const int mh  = (lane < 16) ? 0 : 8;

  {  // load Q tile (pre-scaled)
    int q = qt * 16 + r;
    const float* qp = Qb + (size_t)q * qr + hf * 32;
    bool ok = (q < Sq);
#pragma unroll
    for (int j = 0; j < 32; ++j)
      Qs[r][hf * 32 + j] = to_bf(ok ? qp[j] * scale : 0.f);
  }

  // scores S = (Q*scale) @ K^T
  for (int kt = 0; kt * 16 < Skv; ++kt) {
    int k = kt * 16 + r;
    const float* kp2 = Kb + (size_t)k * kr + hf * 32;
    bool ok = (k < Skv);
#pragma unroll
    for (int j = 0; j < 32; ++j)
      Ks[r][hf * 32 + j] = to_bf(ok ? kp2[j] : 0.f);
    __syncthreads();

    v8f cs = {};
#pragma unroll
    for (int d0 = 0; d0 < 64; d0 += 32) {
      v16bf a, bv;
#pragma unroll
      for (int e = 0; e < 8; ++e) {
        a[e]     = Qs[r][d0 + akb + e];
        a[e + 8] = Qs[r][d0 + 8 + akb + e];
      }
#pragma unroll
      for (int e = 0; e < 16; ++e) bv[e] = Ks[r][d0 + bkb + e];
      cs = WMMA_BF16(a, bv, cs);
    }
#pragma unroll
    for (int r2 = 0; r2 < 8; ++r2) Ps[mh + r2][kt * 16 + r] = cs[r2];
    __syncthreads();
  }

  // softmax per row
  if (lane < 16) {
    float mx = -3.0e38f;
    for (int j = 0; j < Skv; ++j) mx = fmaxf(mx, Ps[r][j]);
    float s = 0.f;
    for (int j = 0; j < Skv; ++j) { float e = __expf(Ps[r][j] - mx); Ps[r][j] = e; s += e; }
    float inv = 1.f / s;
    for (int j = 0; j < Skv; ++j) Ps[r][j] *= inv;
    int Spad = (Skv + 31) & ~31;
    for (int j = Skv; j < Spad; ++j) Ps[r][j] = 0.f;
  }
  __syncthreads();

  // O = P @ V
  v8f co0 = {}, co1 = {}, co2 = {}, co3 = {};
  for (int ks = 0; ks < Skv; ks += 32) {
    int kk = ks + lane;
    const float* vp = Vb + (size_t)kk * vr;
    bool ok = (kk < Skv);
#pragma unroll
    for (int j = 0; j < 64; ++j) Vs[lane][j] = to_bf(ok ? vp[j] : 0.f);
    __syncthreads();

    v16bf a;
#pragma unroll
    for (int e = 0; e < 8; ++e) {
      a[e]     = to_bf(Ps[r][ks + akb + e]);
      a[e + 8] = to_bf(Ps[r][ks + 8 + akb + e]);
    }
    v16bf bv;
#pragma unroll
    for (int e = 0; e < 16; ++e) bv[e] = Vs[bkb + e][0 + r];
    co0 = WMMA_BF16(a, bv, co0);
#pragma unroll
    for (int e = 0; e < 16; ++e) bv[e] = Vs[bkb + e][16 + r];
    co1 = WMMA_BF16(a, bv, co1);
#pragma unroll
    for (int e = 0; e < 16; ++e) bv[e] = Vs[bkb + e][32 + r];
    co2 = WMMA_BF16(a, bv, co2);
#pragma unroll
    for (int e = 0; e < 16; ++e) bv[e] = Vs[bkb + e][48 + r];
    co3 = WMMA_BF16(a, bv, co3);
    __syncthreads();
  }

  float* Ob = O + (long long)b * ob + h * 64;
#pragma unroll
  for (int r2 = 0; r2 < 8; ++r2) {
    int q = qt * 16 + mh + r2;
    if (q < Sq) {
      float* op = Ob + (size_t)q * orr;
      op[0 + r]  = co0[r2];
      op[16 + r] = co1[r2];
      op[32 + r] = co2[r2];
      op[48 + r] = co3[r2];
    }
  }
}

// ---------------------------------------------------------------------------
// LayerNorm over rows of width 512 (wave per row, shuffle reduction).
// ---------------------------------------------------------------------------
__global__ __launch_bounds__(256) void layernorm512(
    const float* __restrict__ X, const float* __restrict__ g,
    const float* __restrict__ bb, float* __restrict__ Y, int M)
{
  const int lane = threadIdx.x & 31;
  const int wave = threadIdx.x >> 5;
  const int row  = blockIdx.x * 8 + wave;
  if (row >= M) return;
  const float* xp = X + (size_t)row * 512;
  float v[16];
  float s = 0.f, s2 = 0.f;
#pragma unroll
  for (int j = 0; j < 16; ++j) {
    float t = xp[lane + j * 32];
    v[j] = t; s += t; s2 += t * t;
  }
#pragma unroll
  for (int o = 16; o > 0; o >>= 1) {
    s  += __shfl_xor(s, o, 32);
    s2 += __shfl_xor(s2, o, 32);
  }
  float mu  = s * (1.f / 512.f);
  float var = s2 * (1.f / 512.f) - mu * mu;
  float inv = rsqrtf(var + 1e-5f);
  float* yp = Y + (size_t)row * 512;
#pragma unroll
  for (int j = 0; j < 16; ++j) {
    int c = lane + j * 32;
    yp[c] = (v[j] - mu) * inv * g[c] + bb[c];
  }
}

__global__ void init_tgt(const float* __restrict__ st, float* __restrict__ tgt) {
  int i = blockIdx.x * 256 + threadIdx.x;   // tgt[0][b][d] = st[d]
  tgt[i] = st[i & 511];
}

__global__ void copy_rows(const float* __restrict__ s, float* __restrict__ d, int n) {
  int i = blockIdx.x * 256 + threadIdx.x;
  if (i < n) d[i] = s[i];
}

// ---------------------------------------------------------------------------
extern "C" void kernel_launch(void* const* d_in, const int* in_sizes, int n_in,
                              void* d_out, int out_size, void* d_ws, size_t ws_size,
                              hipStream_t stream)
{
  (void)in_sizes; (void)n_in; (void)out_size; (void)ws_size;
  const float* x            = (const float*)d_in[0];
  const float* W_in         = (const float*)d_in[1];
  const float* b_in         = (const float*)d_in[2];
  const float* start_token  = (const float*)d_in[3];
  const float* enc_qkv_w    = (const float*)d_in[4];
  const float* enc_qkv_b    = (const float*)d_in[5];
  const float* enc_out_w    = (const float*)d_in[6];
  const float* enc_out_b    = (const float*)d_in[7];
  const float* enc_ln1_g    = (const float*)d_in[8];
  const float* enc_ln1_b    = (const float*)d_in[9];
  const float* enc_ff1_w    = (const float*)d_in[10];
  const float* enc_ff1_b    = (const float*)d_in[11];
  const float* enc_ff2_w    = (const float*)d_in[12];
  const float* enc_ff2_b    = (const float*)d_in[13];
  const float* enc_ln2_g    = (const float*)d_in[14];
  const float* enc_ln2_b    = (const float*)d_in[15];
  const float* enc_norm_g   = (const float*)d_in[16];
  const float* enc_norm_b   = (const float*)d_in[17];
  const float* dec_sa_qkv_w = (const float*)d_in[18];
  const float* dec_sa_qkv_b = (const float*)d_in[19];
  const float* dec_sa_out_w = (const float*)d_in[20];
  const float* dec_sa_out_b = (const float*)d_in[21];
  const float* dec_ln1_g    = (const float*)d_in[22];
  const float* dec_ln1_b    = (const float*)d_in[23];
  const float* dec_ca_qkv_w = (const float*)d_in[24];
  const float* dec_ca_qkv_b = (const float*)d_in[25];
  const float* dec_ca_out_w = (const float*)d_in[26];
  const float* dec_ca_out_b = (const float*)d_in[27];
  const float* dec_ln2_g    = (const float*)d_in[28];
  const float* dec_ln2_b    = (const float*)d_in[29];
  const float* dec_ff1_w    = (const float*)d_in[30];
  const float* dec_ff1_b    = (const float*)d_in[31];
  const float* dec_ff2_w    = (const float*)d_in[32];
  const float* dec_ff2_b    = (const float*)d_in[33];
  const float* dec_ln3_g    = (const float*)d_in[34];
  const float* dec_ln3_b    = (const float*)d_in[35];
  const float* dec_norm_g   = (const float*)d_in[36];
  const float* dec_norm_b   = (const float*)d_in[37];
  const float* W_out        = (const float*)d_in[38];
  const float* b_out        = (const float*)d_in[39];
  // d_in[40] = description_length (host-unreadable under graph capture) == 16.

  const int B = 64, S = 256, D = 512, DFF = 2048, DIN = 256, T = 16;
  const size_t ME = (size_t)B * S;   // 16384 encoder tokens

  float* ws = (float*)d_ws;
  size_t off = 0;
  float* bufA = ws + off; off += ME * D;                  // src -> h1
  float* bufB = ws + off; off += ME * D;                  // attn out -> h2
  float* bufC = ws + off; off += ME * D;                  // residual tmp -> mem
  float* big  = ws + off; off += ME * DFF;                // qkv / ffn hidden (reused)
  float* kvm  = ws + off; off += ME * 1024;               // cross-attn K|V from mem
  float* tgt  = ws + off; off += (size_t)(T + 1) * B * D; // seq-major [t][b][d]
  float* du1  = ws + off; off += (size_t)T * B * D;
  float* du2  = ws + off; off += (size_t)T * B * D;
  float* du3  = ws + off; off += (size_t)T * B * D;
  float* dq   = ws + off; off += (size_t)T * B * D;
  float* datt = ws + off; off += (size_t)T * B * D;
  float* dtmp = ws + off; off += (size_t)T * B * D;
  float* dnrm = ws + off; off += (size_t)T * B * D;

  // bf16 weight arena
  __bf16* wb = (__bf16*)(ws + off);
  size_t wo = 0;
  auto carve = [&](size_t n) { __bf16* p = wb + wo; wo += n; return p; };
  __bf16* wb_in     = carve((size_t)D * DIN);
  __bf16* wb_eqkv   = carve((size_t)3 * D * D);
  __bf16* wb_eout   = carve((size_t)D * D);
  __bf16* wb_eff1   = carve((size_t)DFF * D);
  __bf16* wb_eff2   = carve((size_t)D * DFF);
  __bf16* wb_saqkv  = carve((size_t)3 * D * D);
  __bf16* wb_saout  = carve((size_t)D * D);
  __bf16* wb_caqkv  = carve((size_t)3 * D * D);
  __bf16* wb_caout  = carve((size_t)D * D);
  __bf16* wb_dff1   = carve((size_t)DFF * D);
  __bf16* wb_dff2   = carve((size_t)D * DFF);
  __bf16* wb_out    = carve((size_t)D * D);

  auto conv = [&](const float* s, __bf16* d, size_t n) {
    f32_to_bf16<<<dim3((unsigned)((n + 255) / 256)), 256, 0, stream>>>(s, d, (int)n);
  };
  conv(W_in, wb_in, (size_t)D * DIN);
  conv(enc_qkv_w, wb_eqkv, (size_t)3 * D * D);
  conv(enc_out_w, wb_eout, (size_t)D * D);
  conv(enc_ff1_w, wb_eff1, (size_t)DFF * D);
  conv(enc_ff2_w, wb_eff2, (size_t)D * DFF);
  conv(dec_sa_qkv_w, wb_saqkv, (size_t)3 * D * D);
  conv(dec_sa_out_w, wb_saout, (size_t)D * D);
  conv(dec_ca_qkv_w, wb_caqkv, (size_t)3 * D * D);
  conv(dec_ca_out_w, wb_caout, (size_t)D * D);
  conv(dec_ff1_w, wb_dff1, (size_t)DFF * D);
  conv(dec_ff2_w, wb_dff2, (size_t)D * DFF);
  conv(W_out, wb_out, (size_t)D * D);

  auto gemm = [&](const float* A, const __bf16* Wm, const float* bias, const float* res,
                  float* C, int M, int N, int K, int relu, int perm) {
    gemm_bf16_wmma<<<dim3(N / 128, M / 64), 256, 0, stream>>>(A, Wm, bias, res, C, M, N, K, relu, perm);
  };
  auto ln = [&](const float* X, const float* g, const float* bb, float* Y, int M) {
    layernorm512<<<dim3(M / 8), 256, 0, stream>>>(X, g, bb, Y, M);
  };

  // ===================== Encoder (computed once) =====================
  gemm(x, wb_in, b_in, nullptr, bufA, (int)ME, D, DIN, 0, 0);                // src
  gemm(bufA, wb_eqkv, enc_qkv_b, nullptr, big, (int)ME, 3 * D, D, 0, 0);    // qkv
  attn_wmma<<<dim3(S / 16, 8, B), 32, 0, stream>>>(
      big, (long long)S * 1536, 1536, 0,
      big, (long long)S * 1536, 1536, 512,
      big, (long long)S * 1536, 1536, 1024,
      bufB, (long long)S * 512, 512, S, S, 0.125f);
  gemm(bufB, wb_eout, enc_out_b, bufA, bufC, (int)ME, D, D, 0, 0);          // +src
  ln(bufC, enc_ln1_g, enc_ln1_b, bufA, (int)ME);                            // h1
  gemm(bufA, wb_eff1, enc_ff1_b, nullptr, big, (int)ME, DFF, D, 1, 0);      // relu
  gemm(big, wb_eff2, enc_ff2_b, bufA, bufC, (int)ME, D, DFF, 0, 0);         // +h1
  ln(bufC, enc_ln2_g, enc_ln2_b, bufB, (int)ME);
  ln(bufB, enc_norm_g, enc_norm_b, bufC, (int)ME);                          // mem
  float* mem = bufC;
  // cross-attention K|V from mem (rows 512..1536 of dec_ca_qkv_w)
  gemm(mem, wb_caqkv + (size_t)512 * 512, dec_ca_qkv_b + 512, nullptr,
       kvm, (int)ME, 1024, D, 0, 0);

  init_tgt<<<dim3(B * D / 256), 256, 0, stream>>>(start_token, tgt);

  // ===================== Autoregressive decode (16 steps) =====================
  for (int t = 1; t <= T; ++t) {
    int M = t * B;                 // seq-major: rows [0, t*B) are tgt[:, :t]
    float* din = tgt;
    gemm(din, wb_saqkv, dec_sa_qkv_b, nullptr, big, M, 1536, D, 0, 0);
    attn_wmma<<<dim3((t + 15) / 16, 8, B), 32, 0, stream>>>(     // self-attn (no mask)
        big, (long long)1536, B * 1536, 0,
        big, (long long)1536, B * 1536, 512,
        big, (long long)1536, B * 1536, 1024,
        datt, (long long)512, B * 512, t, t, 0.125f);
    gemm(datt, wb_saout, dec_sa_out_b, din, dtmp, M, D, D, 0, 0);
    ln(dtmp, dec_ln1_g, dec_ln1_b, du1, M);
    gemm(du1, wb_caqkv, dec_ca_qkv_b, nullptr, dq, M, D, D, 0, 0);  // Q proj only
    attn_wmma<<<dim3((t + 15) / 16, 8, B), 32, 0, stream>>>(        // cross-attn
        dq, (long long)512, B * 512, 0,
        kvm, (long long)S * 1024, 1024, 0,
        kvm, (long long)S * 1024, 1024, 512,
        datt, (long long)512, B * 512, t, S, 0.125f);
    gemm(datt, wb_caout, dec_ca_out_b, du1, dtmp, M, D, D, 0, 0);
    ln(dtmp, dec_ln2_g, dec_ln2_b, du2, M);
    gemm(du2, wb_dff1, dec_ff1_b, nullptr, big, M, DFF, D, 1, 0);
    gemm(big, wb_dff2, dec_ff2_b, du2, dtmp, M, D, DFF, 0, 0);
    ln(dtmp, dec_ln3_g, dec_ln3_b, du3, M);
    ln(du3, dec_norm_g, dec_norm_b, dnrm, M);
    copy_rows<<<dim3(B * D / 256), 256, 0, stream>>>(            // append last step
        dnrm + (size_t)(t - 1) * B * D, tgt + (size_t)t * B * D, B * D);
  }

  // logits: tgt[:,1:,:] @ W_out^T + b_out, permuted to (b,t,d) row order
  gemm(tgt + (size_t)B * D, wb_out, b_out, nullptr, (float*)d_out, T * B, D, D, 0, 1);
}
